// Cheb_ZINC_21844203668360
// MI455X (gfx1250) — matrix operand order repaired and checked by reading, not compiled, not run on hardware
//
#include <hip/hip_runtime.h>
#include <hip/hip_bf16.h>

typedef __attribute__((ext_vector_type(16))) __bf16 v16bf;
typedef __attribute__((ext_vector_type(8)))  __bf16 v8bf;
typedef __attribute__((ext_vector_type(4)))  __bf16 v4bf;
typedef __attribute__((ext_vector_type(8)))  float  v8f;

#define THREADS 256
#define KC 256                 // K staged per LDS chunk
#define LDS_STRIDE (KC + 8)    // pad: 528B/col = 132 dwords (≡4 mod 64 banks), 16B aligned

static inline long cdivl(long a, long b) { return (a + b - 1) / b; }
static inline size_t alignUp(size_t x, size_t a) { return (x + a - 1) & ~(a - 1); }

// ---------------- elementwise / setup kernels ----------------

__global__ void zero_kernel(float* __restrict__ p, long n) {
  long i = (long)blockIdx.x * blockDim.x + threadIdx.x;
  if (i < n) p[i] = 0.0f;
}

__global__ void zero_slice_kernel(float* __restrict__ base, int stride, long N, int F) {
  long i = (long)blockIdx.x * blockDim.x + threadIdx.x;
  long total = N * (long)F;
  if (i >= total) return;
  long n = i / F; int f = (int)(i % F);
  base[n * (long)stride + f] = 0.0f;
}

__global__ void degree_kernel(const int* __restrict__ ed, float* __restrict__ deg, long E) {
  long i = (long)blockIdx.x * blockDim.x + threadIdx.x;
  if (i < E) atomicAdd(&deg[ed[i]], 1.0f);
}

__global__ void rsqrt_kernel(const float* __restrict__ deg, float* __restrict__ isd, long N) {
  long i = (long)blockIdx.x * blockDim.x + threadIdx.x;
  if (i < N) isd[i] = rsqrtf(fmaxf(deg[i], 1.0f));
}

// dst[n, :F] = src[idx ? idx[n] : n, :F]
__global__ void gather_rows_kernel(const float* __restrict__ src, int srcStride,
                                   const int* __restrict__ idx,
                                   float* __restrict__ dst, int dstStride,
                                   long N, int F) {
  long i = (long)blockIdx.x * blockDim.x + threadIdx.x;
  long total = N * (long)F;
  if (i >= total) return;
  long n = i / F; int f = (int)(i % F);
  long r = idx ? (long)idx[n] : n;
  dst[n * (long)dstStride + f] = src[r * (long)srcStride + f];
}

// agg[dst, f] += h[src, f] * isd[src]   (edge x float4-chunk parallel)
__global__ void edge_scatter_kernel(const float* __restrict__ hsrc, int srcStride,
                                    float* __restrict__ agg, int dstStride,
                                    const int* __restrict__ es, const int* __restrict__ ed,
                                    const float* __restrict__ isd, long E, int F) {
  long tid = (long)blockIdx.x * blockDim.x + threadIdx.x;
  int chunks = F >> 2;
  long total = E * (long)chunks;
  if (tid >= total) return;
  long e = tid / chunks; int c = (int)(tid % chunks);
  int s = es[e], d = ed[e];
  float w = isd[s];
  const float4 v = *(const float4*)(hsrc + (size_t)s * srcStride + (size_t)c * 4);
  float* p = agg + (size_t)d * dstStride + (size_t)c * 4;
  atomicAdd(p + 0, v.x * w);
  atomicAdd(p + 1, v.y * w);
  atomicAdd(p + 2, v.z * w);
  atomicAdd(p + 3, v.w * w);
}

// t[n,f] = a * t[n,f] * isd[n] + (t0 ? b * t0[n,f] : 0)
__global__ void scale_combine_kernel(float* __restrict__ t, int stride,
                                     const float* __restrict__ t0,
                                     const float* __restrict__ isd,
                                     float a, float b, long N, int F) {
  long i = (long)blockIdx.x * blockDim.x + threadIdx.x;
  long total = N * (long)F;
  if (i >= total) return;
  long n = i / F; int f = (int)(i % F);
  float v = a * t[n * (long)stride + f] * isd[n];
  if (t0) v += b * t0[n * (long)stride + f];
  t[n * (long)stride + f] = v;
}

// hg[gid[n], f] += H[n, f]
__global__ void graph_pool_kernel(const float* __restrict__ H, const int* __restrict__ gid,
                                  float* __restrict__ hg, long N, int F) {
  long tid = (long)blockIdx.x * blockDim.x + threadIdx.x;
  int chunks = F >> 2;
  long total = N * (long)chunks;
  if (tid >= total) return;
  long n = tid / chunks; int c = (int)(tid % chunks);
  int g = gid[n];
  const float4 v = *(const float4*)(H + (size_t)n * F + (size_t)c * 4);
  float* p = hg + (size_t)g * F + (size_t)c * 4;
  atomicAdd(p + 0, v.x);
  atomicAdd(p + 1, v.y);
  atomicAdd(p + 2, v.z);
  atomicAdd(p + 3, v.w);
}

// WT[n, k] = (bf16) W[k, n]
__global__ void transpose_convert_kernel(const float* __restrict__ W, __bf16* __restrict__ WT,
                                         int K, int Nout) {
  long i = (long)blockIdx.x * blockDim.x + threadIdx.x;
  long total = (long)K * Nout;
  if (i >= total) return;
  int k = (int)(i / Nout), n = (int)(i % Nout);
  WT[(size_t)n * K + k] = (__bf16)W[i];
}

// dst[i] = (bf16) src[i], vectorized x4 (n must be a multiple of 4)
__global__ void convert_bf16_kernel(const float* __restrict__ src, __bf16* __restrict__ dst,
                                    long n4) {
  long i = (long)blockIdx.x * blockDim.x + threadIdx.x;
  if (i >= n4) return;
  const float4 v = ((const float4*)src)[i];
  v4bf o;
  o[0] = (__bf16)v.x; o[1] = (__bf16)v.y; o[2] = (__bf16)v.z; o[3] = (__bf16)v.w;
  ((v4bf*)dst)[i] = o;
}

// join two 8-element bf16 vectors (16B-aligned loads) into a fragment
__device__ __forceinline__ v16bf ld16(const __bf16* p) {
  v8bf lo = *(const v8bf*)p;
  v8bf hi = *(const v8bf*)(p + 8);
  return __builtin_shufflevector(lo, hi, 0, 1, 2, 3, 4, 5, 6, 7,
                                 8, 9, 10, 11, 12, 13, 14, 15);
}

// ---------------- WMMA GEMM: C[M,Nout] = act(A[M,K] @ W[K,Nout] + bias) ----------------
// A bf16 [M, lda]; WT = W^T bf16 [Nout, K]. blockDim = 128 (4 waves).
// Block computes 64 rows x 64 cols; wave w owns rows [blockIdx.x*64 + w*16, +16)
// with 4 accumulators over cols [blockIdx.y*64, +64).
// B strip (64 cols x KC) is staged in LDS per K-chunk and shared by all 4 waves.
// Requires: M % 64 == 0, Nout % 64 == 0, K % 32 == 0 (true at all call sites).
// Fragment layouts used by this kernel (wave32):
//   A 16x32 bf16: lanes 0-15 row M=lp; elems 0-15 = K kb..kb+15, kb=kc+k0+half*16
//   B 32x16 bf16: lane lp holds col n; half 0 -> K k0..k0+15, half 1 -> K k0+16..k0+31
//   C 16x16 f32 : VGPR v -> row = mBase + half*8 + v, col = nBase + lp
__global__ void wmma_gemm_kernel(const __bf16* __restrict__ A, int lda,
                                 const __bf16* __restrict__ WT,
                                 const float* __restrict__ bias,
                                 float* __restrict__ C, int ldc,
                                 int K, int Nout, int relu) {
  __shared__ __bf16 ldsB[64 * LDS_STRIDE];   // 33,792 bytes

  const int wave = threadIdx.x >> 5;
  const int lane = threadIdx.x & 31;
  const int half = lane >> 4;
  const int lp   = lane & 15;
  const int mBase = blockIdx.x * 64 + wave * 16;
  const int nBase = blockIdx.y * 64;

  const __bf16* aRow = A + (size_t)(mBase + lp) * lda;

  v8f acc0 = {}, acc1 = {}, acc2 = {}, acc3 = {};

  for (int kc = 0; kc < K; kc += KC) {
    const int kLen = (K - kc < KC) ? (K - kc) : KC;  // 128 or 256, multiple of 32
    __syncthreads();
    // cooperative fill: 64 cols x kLen bf16, 8 elems (16B) per iteration
    const int iters = 64 * (kLen >> 3);
    for (int i = threadIdx.x; i < iters; i += 128) {
      const int col = i / (kLen >> 3);
      const int ko  = (i - col * (kLen >> 3)) << 3;
      *(v8bf*)(&ldsB[col * LDS_STRIDE + ko]) =
          *(const v8bf*)(WT + (size_t)(nBase + col) * K + kc + ko);
    }
    __syncthreads();

    for (int k0 = 0; k0 < kLen; k0 += 32) {
      const int kb = kc + k0 + half * 16;
      const v16bf a = ld16(aRow + kb);     // loads at kb and kb+16 internally? no:
      // A uses contiguous K (kb..kb+15) to match the B fragment:
      const v8bf alo = *(const v8bf*)(aRow + kb);
      const v8bf ahi = *(const v8bf*)(aRow + kb + 8);
      const v16bf af = __builtin_shufflevector(alo, ahi, 0, 1, 2, 3, 4, 5, 6, 7,
                                               8, 9, 10, 11, 12, 13, 14, 15);
      (void)a;

      const int kbB = k0 + half * 16;      // 16 contiguous bf16 in LDS
      const v16bf f0 = ld16(&ldsB[(0 * 16 + lp) * LDS_STRIDE + kbB]);
      const v16bf f1 = ld16(&ldsB[(1 * 16 + lp) * LDS_STRIDE + kbB]);
      const v16bf f2 = ld16(&ldsB[(2 * 16 + lp) * LDS_STRIDE + kbB]);
      const v16bf f3 = ld16(&ldsB[(3 * 16 + lp) * LDS_STRIDE + kbB]);

      acc0 = __builtin_amdgcn_wmma_f32_16x16x32_bf16(false, af, false, f0, (short)0, acc0, false, false);
      acc1 = __builtin_amdgcn_wmma_f32_16x16x32_bf16(false, af, false, f1, (short)0, acc1, false, false);
      acc2 = __builtin_amdgcn_wmma_f32_16x16x32_bf16(false, af, false, f2, (short)0, acc2, false, false);
      acc3 = __builtin_amdgcn_wmma_f32_16x16x32_bf16(false, af, false, f3, (short)0, acc3, false, false);
    }
  }

  const int rowBase = mBase + half * 8;
#pragma unroll
  for (int t = 0; t < 4; ++t) {
    const v8f acc = (t == 0) ? acc0 : (t == 1) ? acc1 : (t == 2) ? acc2 : acc3;
    const int col = nBase + t * 16 + lp;
    const float bv = bias[col];
#pragma unroll
    for (int v = 0; v < 8; ++v) {
      float x = acc[v] + bv;
      if (relu) x = fmaxf(x, 0.0f);
      C[(size_t)(rowBase + v) * ldc + col] = x;
    }
  }
}

// out[g] = dot(m1[g,:], Wm2) + bm2   (one wave32 per graph)
__global__ void mlp2_kernel(const float* __restrict__ m1, const float* __restrict__ Wm2,
                            const float* __restrict__ bm2, float* __restrict__ out,
                            int G, int F) {
  int g = blockIdx.x;
  int lane = threadIdx.x;
  float s = 0.0f;
  for (int f = lane; f < F; f += 32) s += m1[(size_t)g * F + f] * Wm2[f];
#pragma unroll
  for (int off = 16; off > 0; off >>= 1) s += __shfl_down(s, off, 32);
  if (lane == 0) out[g] = s + bm2[0];
}

// ---------------- host-side orchestration ----------------

extern "C" void kernel_launch(void* const* d_in, const int* in_sizes, int n_in,
                              void* d_out, int out_size, void* d_ws, size_t ws_size,
                              hipStream_t stream) {
  const int*   signal = (const int*)d_in[0];
  const int*   es     = (const int*)d_in[1];
  const int*   ed     = (const int*)d_in[2];
  const int*   gid    = (const int*)d_in[3];
  const float* emb    = (const float*)d_in[5];
  const float* Wc1    = (const float*)d_in[6];
  const float* bc1    = (const float*)d_in[7];
  const float* Wc2    = (const float*)d_in[8];
  const float* bc2    = (const float*)d_in[9];
  const float* Wc3    = (const float*)d_in[10];
  const float* bc3    = (const float*)d_in[11];
  const float* Wm1    = (const float*)d_in[12];
  const float* bm1    = (const float*)d_in[13];
  const float* Wm2    = (const float*)d_in[14];
  const float* bm2    = (const float*)d_in[15];

  const long N = in_sizes[0];
  const long E = in_sizes[1];
  const int  G = out_size;        // n_graphs (output is [G,1] f32)
  const int  F1 = 128, F2 = 256, F3 = 512;

  // workspace layout
  char* ws = (char*)d_ws;
  size_t off = 0;
  float*  deg  = (float*)(ws + off);  off = alignUp(off + (size_t)N * 4, 256);
  float*  isd  = (float*)(ws + off);  off = alignUp(off + (size_t)N * 4, 256);
  float*  Tcat = (float*)(ws + off);  off = alignUp(off + (size_t)N * 768 * 4, 256);
  float*  H    = (float*)(ws + off);  off = alignUp(off + (size_t)N * 512 * 4, 256);
  float*  hg   = (float*)(ws + off);  off = alignUp(off + (size_t)G * 512 * 4, 256);
  float*  m1   = (float*)(ws + off);  off = alignUp(off + (size_t)G * 512 * 4, 256);
  __bf16* WT   = (__bf16*)(ws + off); off = alignUp(off + (size_t)768 * 512 * 2, 256);
  __bf16* Abf  = (__bf16*)(ws + off); off = alignUp(off + (size_t)N * 768 * 2, 256);
  __bf16* hgbf = (__bf16*)(ws + off); off = alignUp(off + (size_t)G * 512 * 2, 256);
  (void)ws_size;

  const dim3 blk(THREADS);
  auto grid1 = [](long total) { return dim3((unsigned)cdivl(total, THREADS)); };

  // --- normalization ---
  zero_kernel<<<grid1(N), blk, 0, stream>>>(deg, N);
  degree_kernel<<<grid1(E), blk, 0, stream>>>(ed, deg, E);
  rsqrt_kernel<<<grid1(N), blk, 0, stream>>>(deg, isd, N);

  // Per-layer Chebyshev pipeline, then bf16 WMMA GEMM
  struct Layer { const float *W, *b; int Fin, Fout; };
  for (int layer = 0; layer < 3; ++layer) {
    const int F = (layer == 2) ? F2 : F1;      // input feature width
    const int S = 3 * F;
    const int Kdim = S;
    const int Fout = (layer == 0) ? 128 : (layer == 1) ? 256 : 512;
    const float* W = (layer == 0) ? Wc1 : (layer == 1) ? Wc2 : Wc3;
    const float* b = (layer == 0) ? bc1 : (layer == 1) ? bc2 : bc3;

    if (layer == 0)
      gather_rows_kernel<<<grid1(N * F), blk, 0, stream>>>(emb, F1, signal, Tcat, S, N, F);
    else
      gather_rows_kernel<<<grid1(N * F), blk, 0, stream>>>(H, F, nullptr, Tcat, S, N, F);

    zero_slice_kernel<<<grid1(N * F), blk, 0, stream>>>(Tcat + F, S, N, F);
    edge_scatter_kernel<<<grid1(E * (F / 4)), blk, 0, stream>>>(Tcat, S, Tcat + F, S, es, ed, isd, E, F);
    scale_combine_kernel<<<grid1(N * F), blk, 0, stream>>>(Tcat + F, S, nullptr, isd, -1.0f, 0.0f, N, F);
    zero_slice_kernel<<<grid1(N * F), blk, 0, stream>>>(Tcat + 2 * F, S, N, F);
    edge_scatter_kernel<<<grid1(E * (F / 4)), blk, 0, stream>>>(Tcat + F, S, Tcat + 2 * F, S, es, ed, isd, E, F);
    scale_combine_kernel<<<grid1(N * F), blk, 0, stream>>>(Tcat + 2 * F, S, Tcat, isd, -2.0f, -1.0f, N, F);

    convert_bf16_kernel<<<grid1(N * S / 4), blk, 0, stream>>>(Tcat, Abf, N * S / 4);
    transpose_convert_kernel<<<grid1((long)Kdim * Fout), blk, 0, stream>>>(W, WT, Kdim, Fout);
    wmma_gemm_kernel<<<dim3((unsigned)(N / 64), Fout / 64), dim3(128), 0, stream>>>(
        Abf, S, WT, b, H, Fout, Kdim, Fout, 1);
  }

  // =============== readout + MLP ===============
  zero_kernel<<<grid1((long)G * F3), blk, 0, stream>>>(hg, (long)G * F3);
  graph_pool_kernel<<<grid1(N * (F3 / 4)), blk, 0, stream>>>(H, gid, hg, N, F3);

  convert_bf16_kernel<<<grid1((long)G * 512 / 4), blk, 0, stream>>>(hg, hgbf, (long)G * 512 / 4);
  transpose_convert_kernel<<<grid1(512L * 512), blk, 0, stream>>>(Wm1, WT, 512, 512);
  wmma_gemm_kernel<<<dim3((unsigned)(G / 64), 512 / 64), dim3(128), 0, stream>>>(
      hgbf, 512, WT, bm1, m1, 512, 512, 512, 1);

  mlp2_kernel<<<dim3(G), dim3(32), 0, stream>>>(m1, Wm2, bm2, (float*)d_out, G, 512);
}
